// DIMKT_53961969107220
// MI455X (gfx1250) — compile-verified, matrix operand order checked by reading, not run on hardware
//
#include <hip/hip_runtime.h>
#include <hip/hip_bf16.h>

typedef __attribute__((ext_vector_type(16))) _Float16 v16h;
typedef __attribute__((ext_vector_type(8)))  _Float16 v8h;
typedef __attribute__((ext_vector_type(4)))  _Float16 v4h;
typedef __attribute__((ext_vector_type(8)))  float    v8f;
typedef __attribute__((ext_vector_type(4)))  float    v4f;

union H16 { v16h v; struct { v8h lo, hi; } p; };

#define B_    16
#define L_    100
#define E_    128
#define MAXC_ 128
#define NQ_   1000
#define CODE_STRIDE (NQ_*2 + MAXC_*3)

__device__ __forceinline__
v8f wmma_f16(const H16& a, const H16& b, v8f c) {
  return __builtin_amdgcn_wmma_f32_16x16x32_f16(false, a.v, false, b.v,
                                                (short)0, c, false, false);
}

// branchless activations (v_exp_f32 based; no EXEC-mask branching in hot loops)
__device__ __forceinline__ float sigm(float x) {
  return 1.f / (1.f + __expf(-x));
}
__device__ __forceinline__ float fast_tanh(float x) {
  // tanh(x) = 1 - 2/(exp(2x)+1)
  return 1.f - __fdividef(2.f, __expf(2.f * x) + 1.f);
}

// ---------------------------------------------------------------------------
// Pack a row-major [K][N] f32 weight matrix into WMMA-B fragment order (f16):
// dst[(((nt*KS)+ks)*32 + lane)*16 + i] holds W[k][n] with
//   n = nt*16 + (lane&15)
//   kbase = ks*32 + (lane&16 ? 8 : 0)
//   k = (i<8) ? kbase+i : kbase+16+(i-8)
// Mirrors the ISA 16-bit A-matrix 16x32 per-lane layout (transposed operand).
// ---------------------------------------------------------------------------
__global__ void pack_b_kernel(const float* __restrict__ W, _Float16* __restrict__ dst,
                              int K, int N) {
  int KS = K >> 5;
  int NT = N >> 4;
  int total = NT * KS * 32 * 16;
  for (int idx = blockIdx.x * blockDim.x + threadIdx.x; idx < total;
       idx += gridDim.x * blockDim.x) {
    int i    = idx & 15;
    int lane = (idx >> 4) & 31;
    int rest = idx >> 9;            // nt*KS + ks
    int ks   = rest % KS;
    int nt   = rest / KS;
    int n    = nt * 16 + (lane & 15);
    int kb   = ks * 32 + ((lane & 16) ? 8 : 0);
    int k    = (i < 8) ? (kb + i) : (kb + 16 + (i - 8));
    dst[idx] = (_Float16)W[(size_t)k * N + n];
  }
}

// ---------------------------------------------------------------------------
// Fused: gather full[b,l,:, :384] (f16, LDS), tr = tanh(full@Wpt+bpt),
// ctx[b,l,m] = tr . Wat + bat.  One block per (b,l); 2 passes of 64 path rows.
// 8 waves: wave w -> m_tile (w>>1), n-half (w&1) covering 12 of 24 n-tiles.
// Dual accumulators (even/odd k-step) to break WMMA->WMMA RAW hazard chains.
// ---------------------------------------------------------------------------
#define AROW 392   // 384 + 8 f16 pad: 784B row stride, 16B aligned, bank-skewed

__global__ __launch_bounds__(256)
void ctx_kernel(const int* __restrict__ code,
                const float* __restrict__ embed_nodes,
                const float* __restrict__ embed_paths,
                const _Float16* __restrict__ WptP,
                const float* __restrict__ bpt,
                const float* __restrict__ Wat,
                const float* __restrict__ bat,
                float* __restrict__ ctx) {
  __shared__ _Float16 Afull[64 * AROW];
  __shared__ int      idx[MAXC_ * 3];
  __shared__ float    cbuf[2 * 64];

  const int bl  = blockIdx.x;       // b*L + l
  const int tid = threadIdx.x;
  const int wave = tid >> 5, lane = tid & 31;
  const int mt   = wave >> 1;       // local m-tile 0..3
  const int nh   = wave & 1;        // n half
  const int kadd = (lane & 16) ? 8 : 0;
  const float bat0 = bat[0];

  const int* cbase = code + (size_t)bl * CODE_STRIDE + NQ_ * 2;
  for (int t = tid; t < MAXC_ * 3; t += 256) idx[t] = cbase[t];
  __syncthreads();

  for (int pass = 0; pass < 2; ++pass) {
    // ---- gather 64 rows of `full` into LDS as f16 (vectorized x4) ----
    for (int e = tid; e < 64 * 96; e += 256) {
      int ml = e / 96, jq = (e - ml * 96) * 4;
      int m  = pass * 64 + ml;
      const float* src;
      if (jq < 128)      src = embed_nodes + (size_t)idx[m*3 + 0] * E_ + jq;
      else if (jq < 256) src = embed_nodes + (size_t)idx[m*3 + 2] * E_ + (jq - 128);
      else               src = embed_paths + (size_t)idx[m*3 + 1] * E_ + (jq - 256);
      v4f v = *(const v4f*)src;
      *(v4h*)(Afull + ml * AROW + jq) = __builtin_convertvector(v, v4h);
    }
    __syncthreads();

    const _Float16* arow = Afull + (mt * 16 + (lane & 15)) * AROW + kadd;
    float ctxacc[8];
#pragma unroll
    for (int r = 0; r < 8; ++r) ctxacc[r] = 0.f;

    for (int ni = 0; ni < 12; ++ni) {
      int nt = nh * 12 + ni;
      v8f acc0 = {}, acc1 = {};
      const _Float16* bp = WptP + (unsigned)((nt * 12) * 32 + lane) * 16u;
#pragma unroll
      for (int ks = 0; ks < 12; ks += 2) {
        H16 a0, b0, a1, b1;
        const _Float16* ap = arow + ks * 32;
        a0.p.lo = *(const v8h*)(ap);
        a0.p.hi = *(const v8h*)(ap + 16);
        b0.p.lo = *(const v8h*)(bp);
        b0.p.hi = *(const v8h*)(bp + 8);
        a1.p.lo = *(const v8h*)(ap + 32);
        a1.p.hi = *(const v8h*)(ap + 48);
        b1.p.lo = *(const v8h*)(bp + 512);
        b1.p.hi = *(const v8h*)(bp + 520);
        acc0 = wmma_f16(a0, b0, acc0);
        acc1 = wmma_f16(a1, b1, acc1);
        bp += 1024;
      }
      int n = nt * 16 + (lane & 15);
      float bp0 = bpt[n], wa = Wat[n];
#pragma unroll
      for (int r = 0; r < 8; ++r) {
        float t = fast_tanh(acc0[r] + acc1[r] + bp0);
        ctxacc[r] = __builtin_fmaf(t, wa, ctxacc[r]);
      }
    }
    // reduce over the 16 lanes of each half (N dimension)
#pragma unroll
    for (int r = 0; r < 8; ++r) {
      float v = ctxacc[r];
      v += __shfl_xor(v, 1, 32);
      v += __shfl_xor(v, 2, 32);
      v += __shfl_xor(v, 4, 32);
      v += __shfl_xor(v, 8, 32);
      ctxacc[r] = v;
    }
    if ((lane & 15) == 0) {
      int mbase = mt * 16 + ((lane & 16) ? 8 : 0);
#pragma unroll
      for (int r = 0; r < 8; ++r)
        cbuf[nh * 64 + mbase + r] = ctxacc[r];
    }
    __syncthreads();
    if (tid < 64)
      ctx[(size_t)bl * MAXC_ + pass * 64 + tid] = cbuf[tid] + cbuf[64 + tid] + bat0;
    __syncthreads();
  }
}

// ---------------------------------------------------------------------------
// Softmax over the sequence axis (dim=1), per (b, path m).  Coalesced passes.
// ---------------------------------------------------------------------------
__global__ __launch_bounds__(128)
void softmax_kernel(const float* __restrict__ ctx, float* __restrict__ attn) {
  int b = blockIdx.x, m = threadIdx.x;
  const float* p = ctx + (size_t)b * L_ * MAXC_ + m;
  float mx = -1e30f;
  for (int l = 0; l < L_; ++l) mx = fmaxf(mx, p[l * MAXC_]);
  float s = 0.f;
  for (int l = 0; l < L_; ++l) s += __expf(p[l * MAXC_] - mx);
  float inv = 1.f / s;
  float* q = attn + (size_t)b * L_ * MAXC_ + m;
  for (int l = 0; l < L_; ++l) q[l * MAXC_] = __expf(p[l * MAXC_] - mx) * inv;
}

// ---------------------------------------------------------------------------
// code_vec[b,l,:] = sum_m full[b,l,m,:] * attn[b,l,m]  (re-gather; tables in L2)
// ---------------------------------------------------------------------------
__global__ __launch_bounds__(128)
void codevec_kernel(const int* __restrict__ code,
                    const float* __restrict__ embed_nodes,
                    const float* __restrict__ embed_paths,
                    const float* __restrict__ attn,
                    float* __restrict__ code_vec) {
  __shared__ int   idx[MAXC_ * 3];
  __shared__ float at[MAXC_];
  int bl = blockIdx.x, t = threadIdx.x;
  const int* cbase = code + (size_t)bl * CODE_STRIDE + NQ_ * 2;
  for (int i = t; i < MAXC_ * 3; i += 128) idx[i] = cbase[i];
  at[t] = attn[(size_t)bl * MAXC_ + t];
  __syncthreads();
  float s0 = 0.f, s1 = 0.f, s2 = 0.f;
  for (int m = 0; m < MAXC_; ++m) {
    float a = at[m];
    s0 = __builtin_fmaf(embed_nodes[(size_t)idx[m*3 + 0] * E_ + t], a, s0);
    s1 = __builtin_fmaf(embed_nodes[(size_t)idx[m*3 + 2] * E_ + t], a, s1);
    s2 = __builtin_fmaf(embed_paths[(size_t)idx[m*3 + 1] * E_ + t], a, s2);
  }
  float* o = code_vec + (size_t)bl * 384;
  o[t] = s0; o[128 + t] = s1; o[256 + t] = s2;
}

// ---------------------------------------------------------------------------
// acode = [a_tab[a], code_vec] @ Wqa + bqa;  inp = [q,c,sd,qd embeds] @ W1 + b1
// ---------------------------------------------------------------------------
__global__ __launch_bounds__(128)
void proj_kernel(const int* __restrict__ q, const int* __restrict__ c,
                 const int* __restrict__ sd, const int* __restrict__ qd,
                 const int* __restrict__ a,
                 const float* __restrict__ q_tab, const float* __restrict__ c_tab,
                 const float* __restrict__ sd_tab, const float* __restrict__ qd_tab,
                 const float* __restrict__ a_tab, const float* __restrict__ code_vec,
                 const float* __restrict__ Wqa, const float* __restrict__ bqa,
                 const float* __restrict__ W1, const float* __restrict__ b1,
                 float* __restrict__ acode, float* __restrict__ inp) {
  __shared__ float xa[512];
  __shared__ float xi[512];
  int bl = blockIdx.x, t = threadIdx.x;
  xa[t]       = a_tab[(size_t)a[bl] * E_ + t];
  xa[128 + t] = code_vec[(size_t)bl * 384 + t];
  xa[256 + t] = code_vec[(size_t)bl * 384 + 128 + t];
  xa[384 + t] = code_vec[(size_t)bl * 384 + 256 + t];
  xi[t]       = q_tab[(size_t)q[bl]  * E_ + t];
  xi[128 + t] = c_tab[(size_t)c[bl]  * E_ + t];
  xi[256 + t] = sd_tab[(size_t)sd[bl] * E_ + t];
  xi[384 + t] = qd_tab[(size_t)qd[bl] * E_ + t];
  __syncthreads();
  float sa = bqa[t], si = b1[t];
  for (int i = 0; i < 512; ++i) {
    sa = __builtin_fmaf(xa[i], Wqa[(size_t)i * E_ + t], sa);
    si = __builtin_fmaf(xi[i], W1[(size_t)i * E_ + t], si);
  }
  acode[(size_t)bl * E_ + t] = sa;
  inp[(size_t)bl * E_ + t]   = si;
}

// ---------------------------------------------------------------------------
// Sequential SDF/PKA/KSU recurrence.  1 block, 256 threads (8 waves).
// All recurrence GEMMs as WMMA (M=16 batch rows); wave w owns N-tile w.
// ---------------------------------------------------------------------------
#define QQS  136   // 128+8 f16
#define XBS  264   // 256+8 f16
#define INSS 520   // 512+8 f16

__device__ __forceinline__
v8f gemm_tile(const _Float16* Asm, int astride, const _Float16* Bp,
              int KS, int wave, int lane) {
  v8f acc0 = {}, acc1 = {};
  const _Float16* arow = Asm + (lane & 15) * astride + ((lane & 16) ? 8 : 0);
  const _Float16* bp = Bp + (unsigned)(wave * KS * 512 + lane * 16);
  for (int ks = 0; ks < KS; ks += 2) {
    H16 a0, b0, a1, b1;
    const _Float16* ap = arow + ks * 32;
    a0.p.lo = *(const v8h*)(ap);
    a0.p.hi = *(const v8h*)(ap + 16);
    b0.p.lo = *(const v8h*)(bp);
    b0.p.hi = *(const v8h*)(bp + 8);
    a1.p.lo = *(const v8h*)(ap + 32);
    a1.p.hi = *(const v8h*)(ap + 48);
    b1.p.lo = *(const v8h*)(bp + 512);
    b1.p.hi = *(const v8h*)(bp + 520);
    acc0 = wmma_f16(a0, b0, acc0);
    acc1 = wmma_f16(a1, b1, acc1);
    bp += 1024;
  }
  return acc0 + acc1;
}

__global__ __launch_bounds__(256)
void scan_kernel(const float* __restrict__ inp, const float* __restrict__ acode,
                 const int* __restrict__ sd, const int* __restrict__ qd,
                 const float* __restrict__ sd_tab, const float* __restrict__ qd_tab,
                 const float* __restrict__ knowledge,
                 const _Float16* __restrict__ W2p, const float* __restrict__ b2,
                 const _Float16* __restrict__ W3p, const float* __restrict__ b3,
                 const _Float16* __restrict__ W4p, const float* __restrict__ b4,
                 const _Float16* __restrict__ W5p, const float* __restrict__ b5,
                 const _Float16* __restrict__ W6p, const float* __restrict__ b6,
                 const float* __restrict__ Wf, const float* __restrict__ bf,
                 float* __restrict__ out) {
  __shared__ float    kf[B_ * E_];        // f32 knowledge state
  __shared__ _Float16 qq[B_ * QQS];
  __shared__ _Float16 xb[B_ * XBS];       // [sdf | acode]
  __shared__ _Float16 ins[B_ * INSS];     // [k | acode | sd_e | qd_e]

  const int t = threadIdx.x;
  const int wave = t >> 5, lane = t & 31;
  const int n = wave * 16 + (lane & 15);
  const int moff = (lane & 16) ? 8 : 0;

  for (int e = t; e < B_ * E_; e += 256) kf[e] = knowledge[e & (E_ - 1)];
  __syncthreads();

  const float b2v = b2[n], b3v = b3[n], b4v = b4[n], b5v = b5[n], b6v = b6[n];

  for (int l = 0; l < L_; ++l) {
    // ---- P1: stage qq, ins, acode part of xb (vectorized x4) ----
    for (int eq = t; eq < B_ * 32; eq += 256) {
      int bb = eq >> 5, j = (eq & 31) * 4;
      size_t g = ((size_t)bb * L_ + l) * E_ + j;
      v4f kv = *(const v4f*)(kf + bb * E_ + j);
      v4f iv = *(const v4f*)(inp + g);
      *(v4h*)(qq + bb * QQS + j)         = __builtin_convertvector(iv - kv, v4h);
      *(v4h*)(ins + bb * INSS + j)       = __builtin_convertvector(kv, v4h);
      v4h ach = __builtin_convertvector(*(const v4f*)(acode + g), v4h);
      *(v4h*)(ins + bb * INSS + 128 + j) = ach;
      *(v4h*)(xb + bb * XBS + 128 + j)   = ach;
      v4f sv = *(const v4f*)(sd_tab + (size_t)sd[bb * L_ + l] * E_ + j);
      *(v4h*)(ins + bb * INSS + 256 + j) = __builtin_convertvector(sv, v4h);
      v4f qv = *(const v4f*)(qd_tab + (size_t)qd[bb * L_ + l] * E_ + j);
      *(v4h*)(ins + bb * INSS + 384 + j) = __builtin_convertvector(qv, v4h);
    }
    __syncthreads();

    // ---- P2: sdf = sigmoid(qq@W2+b2) * tanh(qq@W3+b3) ----
    {
      v8f c2 = gemm_tile(qq, QQS, W2p, 4, wave, lane);
      v8f c3 = gemm_tile(qq, QQS, W3p, 4, wave, lane);
#pragma unroll
      for (int r = 0; r < 8; ++r) {
        float v = sigm(c2[r] + b2v) * fast_tanh(c3[r] + b3v);
        xb[(moff + r) * XBS + n] = (_Float16)v;
      }
    }
    __syncthreads();

    // ---- P3: pka, gate, k_new ----
    float knew[8];
    {
      v8f c4 = gemm_tile(xb, XBS, W4p, 8, wave, lane);
      v8f c5 = gemm_tile(xb, XBS, W5p, 8, wave, lane);
      v8f c6 = gemm_tile(ins, INSS, W6p, 16, wave, lane);
#pragma unroll
      for (int r = 0; r < 8; ++r) {
        float pka = sigm(c4[r] + b4v) * fast_tanh(c5[r] + b5v);
        float g   = sigm(c6[r] + b6v);
        float ko  = kf[(moff + r) * E_ + n];
        knew[r] = g * ko + (1.f - g) * pka;
      }
    }
    __syncthreads();   // all reads of kf/xb/ins complete

    // ---- P4: commit state ----
#pragma unroll
    for (int r = 0; r < 8; ++r) kf[(moff + r) * E_ + n] = knew[r];
    __syncthreads();

    // ---- P5: logits = sigmoid(k_new @ Wf + bf) ----
    if (t < B_ * 10) {
      int bb = t / 10, col = t - bb * 10;
      float s = bf[col];
      const float* kr = &kf[bb * E_];
      for (int e = 0; e < E_; ++e)
        s = __builtin_fmaf(kr[e], Wf[e * 10 + col], s);
      out[((size_t)bb * L_ + l) * 10 + col] = sigm(s);
    }
    __syncthreads();
  }
}

// ---------------------------------------------------------------------------
extern "C" void kernel_launch(void* const* d_in, const int* in_sizes, int n_in,
                              void* d_out, int out_size, void* d_ws, size_t ws_size,
                              hipStream_t stream) {
  (void)in_sizes; (void)n_in; (void)out_size; (void)ws_size;
  const int*   code        = (const int*)  d_in[0];
  const int*   q           = (const int*)  d_in[1];
  const int*   c           = (const int*)  d_in[2];
  const int*   sd          = (const int*)  d_in[3];
  const int*   qd          = (const int*)  d_in[4];
  const int*   a           = (const int*)  d_in[5];
  const float* embed_nodes = (const float*)d_in[10];
  const float* embed_paths = (const float*)d_in[11];
  const float* q_tab       = (const float*)d_in[12];
  const float* c_tab       = (const float*)d_in[13];
  const float* sd_tab      = (const float*)d_in[14];
  const float* qd_tab      = (const float*)d_in[15];
  const float* a_tab       = (const float*)d_in[16];
  const float* knowledge   = (const float*)d_in[17];
  const float* Wpt = (const float*)d_in[18]; const float* bpt = (const float*)d_in[19];
  const float* Wat = (const float*)d_in[20]; const float* bat = (const float*)d_in[21];
  const float* Wqa = (const float*)d_in[22]; const float* bqa = (const float*)d_in[23];
  const float* W1  = (const float*)d_in[24]; const float* b1  = (const float*)d_in[25];
  const float* W2  = (const float*)d_in[26]; const float* b2  = (const float*)d_in[27];
  const float* W3  = (const float*)d_in[28]; const float* b3  = (const float*)d_in[29];
  const float* W4  = (const float*)d_in[30]; const float* b4  = (const float*)d_in[31];
  const float* W5  = (const float*)d_in[32]; const float* b5  = (const float*)d_in[33];
  const float* W6  = (const float*)d_in[34]; const float* b6  = (const float*)d_in[35];
  const float* Wf  = (const float*)d_in[36]; const float* bf  = (const float*)d_in[37];
  float* out = (float*)d_out;

  char* ws = (char*)d_ws;
  _Float16* WptP = (_Float16*)(ws + 0);            // 294,912 B
  _Float16* W2p  = (_Float16*)(ws + 294912);       //  32,768 B
  _Float16* W3p  = (_Float16*)(ws + 327680);       //  32,768 B
  _Float16* W4p  = (_Float16*)(ws + 360448);       //  65,536 B
  _Float16* W5p  = (_Float16*)(ws + 425984);       //  65,536 B
  _Float16* W6p  = (_Float16*)(ws + 491520);       // 131,072 B
  float* ctx      = (float*)(ws + 622592);         // 819,200 B
  float* attn     = (float*)(ws + 1441792);        // 819,200 B
  float* code_vec = (float*)(ws + 2260992);        // 2,457,600 B
  float* acode    = (float*)(ws + 4718592);        // 819,200 B
  float* inp      = (float*)(ws + 5537792);        // 819,200 B

  // weight repack (f32 -> WMMA-B f16 fragments)
  pack_b_kernel<<<576, 256, 0, stream>>>(Wpt, WptP, 384, 384);
  pack_b_kernel<<<64,  256, 0, stream>>>(W2,  W2p,  128, 128);
  pack_b_kernel<<<64,  256, 0, stream>>>(W3,  W3p,  128, 128);
  pack_b_kernel<<<128, 256, 0, stream>>>(W4,  W4p,  256, 128);
  pack_b_kernel<<<128, 256, 0, stream>>>(W5,  W5p,  256, 128);
  pack_b_kernel<<<256, 256, 0, stream>>>(W6,  W6p,  512, 128);

  ctx_kernel<<<B_ * L_, 256, 0, stream>>>(code, embed_nodes, embed_paths, WptP,
                                          bpt, Wat, bat, ctx);
  softmax_kernel<<<B_, 128, 0, stream>>>(ctx, attn);
  codevec_kernel<<<B_ * L_, 128, 0, stream>>>(code, embed_nodes, embed_paths,
                                              attn, code_vec);
  proj_kernel<<<B_ * L_, 128, 0, stream>>>(q, c, sd, qd, a, q_tab, c_tab, sd_tab,
                                           qd_tab, a_tab, code_vec, Wqa, bqa,
                                           W1, b1, acode, inp);
  scan_kernel<<<1, 256, 0, stream>>>(inp, acode, sd, qd, sd_tab, qd_tab, knowledge,
                                     W2p, b2, W3p, b3, W4p, b4, W5p, b5, W6p, b6,
                                     Wf, bf, out);
}
